// CinemaTransformV1_3b_81973745811943
// MI455X (gfx1250) — matrix operand order repaired and checked by reading, not compiled, not run on hardware
//
#include <hip/hip_runtime.h>
#include <hip/hip_bf16.h>

// ---------------------------------------------------------------------------
// CinemaTransform on gfx1250 (MI455X): pointwise+LUT kernel, WMMA-f16 implicit
// GEMM conv3x3 layers (BN folded), composite epilogue.
// Conv stack (~67 GFLOP) dominates -> v_wmma_f32_16x16x32_f16.
// Tile staging uses gfx1250 async global->LDS DMA (ASYNCcnt) instead of the
// VGPR round-trip: global_load_async_to_lds_b128 + s_wait_asynccnt.
// ---------------------------------------------------------------------------

typedef __attribute__((ext_vector_type(16))) _Float16 v16h;
typedef __attribute__((ext_vector_type(8)))  _Float16 v8h;
typedef __attribute__((ext_vector_type(8)))  float    v8f;

#define H_IMG 1080
#define W_IMG 1920
#define NPIX  (H_IMG * W_IMG)
#define LUT_L 33

static __device__ __forceinline__ float clampf(float x, float lo, float hi) {
    return fminf(fmaxf(x, lo), hi);
}

// Async DMA one 16-byte chunk global -> LDS (gfx1250, tracked by ASYNCcnt).
// lds_off: wave-relative LDS byte offset (low 32 bits of generic shared addr).
static __device__ __forceinline__ void async_copy16(uint32_t lds_off, const void* gsrc) {
    asm volatile("global_load_async_to_lds_b128 %0, %1, off"
                 :: "v"(lds_off), "v"((unsigned long long)(uintptr_t)gsrc)
                 : "memory");
}
static __device__ __forceinline__ void async_wait0() {
    asm volatile("s_wait_asynccnt 0" ::: "memory");
}

// ---------------------------------------------------------------------------
// BN fold: sc = g*rsqrt(v+eps), bias = b - m*sc
// ---------------------------------------------------------------------------
__global__ void k_fold_bn(const float* __restrict__ g, const float* __restrict__ b,
                          const float* __restrict__ m, const float* __restrict__ v,
                          float* __restrict__ sc, float* __restrict__ bias,
                          int C, int CPAD) {
    int i = blockIdx.x * blockDim.x + threadIdx.x;
    if (i >= CPAD) return;
    if (i < C) {
        float s = g[i] * rsqrtf(v[i] + 1e-5f);
        sc[i] = s;
        bias[i] = b[i] - m[i] * s;
    } else {
        sc[i] = 0.f;
        bias[i] = 0.f;
    }
}

__global__ void k_final_bias(const float* __restrict__ res_b,
                             const float* __restrict__ attn_b,
                             float* __restrict__ bias) {
    int i = threadIdx.x;
    if (i >= 16) return;
    bias[i] = (i < 3) ? res_b[i] : (i == 3 ? attn_b[0] : 0.f);
}

// ---------------------------------------------------------------------------
// Pack conv weights into WMMA B-fragment records.
// Record layout: rec = (nt*3 + dy)*KK + kk ; 512 f16 per record.
// Lane l (l = e>>4) holds column N = l&15, K_local = (l>>4)*16 + j (j = e&15).
// Global K = kk*32 + K_local -> (dx = K/CPACK, c = K%CPACK) for row-group dy.
// BN scale folded into the weight.
// ---------------------------------------------------------------------------
__global__ void k_pack_w(const float* __restrict__ w, const float* __restrict__ sc,
                         _Float16* __restrict__ wpk,
                         int Cin, int Cout, int CPACK, int NT, int KK) {
    int idx = blockIdx.x * blockDim.x + threadIdx.x;
    int NRE = NT * 3 * KK * 512;
    if (idx >= NRE) return;
    int rec = idx >> 9, e = idx & 511;
    int lane = e >> 4, j = e & 15;
    int nt = rec / (3 * KK);
    int rem = rec % (3 * KK);
    int dy = rem / KK, kk = rem % KK;
    int kl = ((lane >> 4) << 4) + j;
    int k = kk * 32 + kl;
    int dx = k / CPACK, c = k % CPACK;
    int o = nt * 16 + (lane & 15);
    float val = 0.f;
    if (dx < 3 && c < Cin && o < Cout) {
        float s = sc ? sc[o] : 1.f;
        val = w[((o * Cin + c) * 3 + dy) * 3 + dx] * s;
    }
    wpk[idx] = (_Float16)val;
}

// Final layer: 3 residual channels + 1 attention channel packed into N=0..3.
__global__ void k_pack_w_final(const float* __restrict__ wres,
                               const float* __restrict__ wattn,
                               _Float16* __restrict__ wpk) {
    const int CPACK = 16, KK = 2;
    int idx = blockIdx.x * blockDim.x + threadIdx.x;
    int NRE = 1 * 3 * KK * 512;
    if (idx >= NRE) return;
    int rec = idx >> 9, e = idx & 511;
    int lane = e >> 4, j = e & 15;
    int rem = rec % (3 * KK);
    int dy = rem / KK, kk = rem % KK;
    int kl = ((lane >> 4) << 4) + j;
    int k = kk * 32 + kl;
    int dx = k / CPACK, c = k % CPACK;
    int o = lane & 15;
    float val = 0.f;
    if (dx < 3 && c < 16) {
        if (o < 3)       val = wres[((o * 16 + c) * 3 + dy) * 3 + dx];
        else if (o == 3) val = wattn[((0 * 16 + c) * 3 + dy) * 3 + dx];
    }
    wpk[idx] = (_Float16)val;
}

// ---------------------------------------------------------------------------
// Pointwise color pipeline + 3D LUT trilinear + blend.
// Writes: xb (float4/pixel, rgb_) and xin (8 f16/pixel: x rgb, xb rgb, 0, 0).
// ---------------------------------------------------------------------------
__global__ void k_pointwise_lut(const float* __restrict__ x,
                                const float* __restrict__ lut,
                                const float* __restrict__ cm,
                                const float* __restrict__ cbias,
                                const float* __restrict__ sh_,
                                const float* __restrict__ mi_,
                                const float* __restrict__ hi_,
                                const float* __restrict__ co_,
                                const float* __restrict__ sat_,
                                const float* __restrict__ wa_,
                                const float* __restrict__ lb_,
                                float4* __restrict__ xb_out,
                                v8h* __restrict__ xin_out) {
    int p = blockIdx.x * blockDim.x + threadIdx.x;
    if (p >= NPIX) return;
    float r0 = x[p], g0 = x[NPIX + p], b0 = x[2 * NPIX + p];

    float M[9];
#pragma unroll
    for (int i = 0; i < 9; ++i) M[i] = clampf(cm[i], 0.9f, 1.1f);
    float bi[3];
#pragma unroll
    for (int i = 0; i < 3; ++i) bi[i] = clampf(cbias[i], -0.02f, 0.02f);

    float xc[3];
    xc[0] = M[0] * r0 + M[1] * g0 + M[2] * b0 + bi[0];
    xc[1] = M[3] * r0 + M[4] * g0 + M[5] * b0 + bi[1];
    xc[2] = M[6] * r0 + M[7] * g0 + M[8] * b0 + bi[2];

    float sh = clampf(sh_[0], -0.02f, 0.05f);
    float mi = clampf(mi_[0], 0.95f, 1.05f);
    float hi = clampf(hi_[0], 0.95f, 1.05f);
#pragma unroll
    for (int i = 0; i < 3; ++i) {
        float v = xc[i];
        float sm = clampf(1.f - v, 0.f, 1.f);
        sm = sm * sm * sm;
        float xs = v + sh * sm * (1.f - v) * 0.5f;
        float xm = powf(clampf(xs, 1e-7f, 1.f), 1.f / mi);
        float hm = clampf(xm, 0.f, 1.f);
        hm = hm * hm * hm;
        xc[i] = clampf(xm * (1.f - hm * (1.f - hi) * 0.5f), 0.f, 1.f);
    }

    float co = clampf(co_[0], 0.98f, 1.05f);
    float xg[3];
#pragma unroll
    for (int i = 0; i < 3; ++i)
        xg[i] = powf(clampf(xc[i], 1e-7f, 1.f), 1.f / co);

    float luma = 0.299f * xg[0] + 0.587f * xg[1] + 0.114f * xg[2];
    float sat = clampf(sat_[0], 0.95f, 1.3f);
    float s0 = luma + sat * (xg[0] - luma);
    float s1 = luma + sat * (xg[1] - luma);
    float s2 = luma + sat * (xg[2] - luma);
    float wa = clampf(wa_[0], -0.02f, 0.05f);
    float cr = clampf(s0 * (1.f + wa), 0.f, 1.f);
    float cg = clampf(s1 * (1.f + wa * 0.3f), 0.f, 1.f);
    float cb = clampf(s2 * (1.f - wa * 0.5f), 0.f, 1.f);

    // --- 3D LUT trilinear ---
    float cx = cr * (LUT_L - 1), cy = cg * (LUT_L - 1), cz = cb * (LUT_L - 1);
    float fx = cx - floorf(cx), fy = cy - floorf(cy), fz = cz - floorf(cz);
    int x0 = min(max((int)floorf(cx), 0), LUT_L - 2);
    int y0 = min(max((int)floorf(cy), 0), LUT_L - 2);
    int z0 = min(max((int)floorf(cz), 0), LUT_L - 2);
    int x1 = x0 + 1, y1 = y0 + 1, z1 = z0 + 1;

    const float* c000 = lut + ((x0 * LUT_L + y0) * LUT_L + z0) * 3;
    const float* c001 = lut + ((x0 * LUT_L + y0) * LUT_L + z1) * 3;
    const float* c010 = lut + ((x0 * LUT_L + y1) * LUT_L + z0) * 3;
    const float* c011 = lut + ((x0 * LUT_L + y1) * LUT_L + z1) * 3;
    const float* c100 = lut + ((x1 * LUT_L + y0) * LUT_L + z0) * 3;
    const float* c101 = lut + ((x1 * LUT_L + y0) * LUT_L + z1) * 3;
    const float* c110 = lut + ((x1 * LUT_L + y1) * LUT_L + z0) * 3;
    const float* c111 = lut + ((x1 * LUT_L + y1) * LUT_L + z1) * 3;

    float lb = clampf(lb_[0], 0.7f, 0.9f);
    float lr[3];
#pragma unroll
    for (int c = 0; c < 3; ++c) {
        float c00 = c000[c] * (1.f - fx) + c100[c] * fx;
        float c01 = c001[c] * (1.f - fx) + c101[c] * fx;
        float c10 = c010[c] * (1.f - fx) + c110[c] * fx;
        float c11 = c011[c] * (1.f - fx) + c111[c] * fx;
        float e0 = c00 * (1.f - fy) + c10 * fy;
        float e1 = c01 * (1.f - fy) + c11 * fy;
        lr[c] = e0 * (1.f - fz) + e1 * fz;
    }
    float xbr = lb * lr[0] + (1.f - lb) * cr;
    float xbg = lb * lr[1] + (1.f - lb) * cg;
    float xbb = lb * lr[2] + (1.f - lb) * cb;

    xb_out[p] = make_float4(xbr, xbg, xbb, 0.f);

    union { v8h v; _Float16 h[8]; } u;
    u.h[0] = (_Float16)r0;  u.h[1] = (_Float16)g0;  u.h[2] = (_Float16)b0;
    u.h[3] = (_Float16)xbr; u.h[4] = (_Float16)xbg; u.h[5] = (_Float16)xbb;
    u.h[6] = (_Float16)0.f; u.h[7] = (_Float16)0.f;
    xin_out[p] = u.v;
}

// ---------------------------------------------------------------------------
// WMMA conv3x3 (SAME, zero pad). Channel-packed f16 features [pixel*CPACK+c].
// Tile: 32x4 output pixels / block (8 waves, 16-pixel M-tile per wave).
// Staging: async global->LDS DMA (no VGPR round-trip); interior tiles take a
// branch-free pure-DMA path, border tiles mask OOB lanes with LDS zero stores.
// For each row-group dy, the 3 x-taps are contiguous K (3*CPACK f16), walked
// in K=32 chunks with v_wmma_f32_16x16x32_f16; K padding carries zero weights.
// Accumulators start at folded-BN bias; optional ReLU; f16 store (COUTS chans).
// ---------------------------------------------------------------------------
template <int CPACK, int NT, int KK, int COUTS, bool RELU>
__global__ __launch_bounds__(256) void k_conv3x3_wmma(
    const _Float16* __restrict__ fin,
    const _Float16* __restrict__ wpk,
    const float* __restrict__ bias,
    _Float16* __restrict__ fout) {
    constexpr int TW = 32, TH = 4;
    constexpr int ROWPIX = 36;                 // staged pixels per row (halo + overread)
    constexpr int NRECS = NT * 3 * KK;
    constexpr int CPP = CPACK / 8;             // 16B chunks per pixel
    constexpr int NCHUNK = 6 * ROWPIX * CPP;
    __shared__ __align__(16) _Float16 sFeat[6 * ROWPIX * CPACK];
    __shared__ __align__(16) _Float16 sW[NRECS * 512];

    const int tid = threadIdx.x;
    const int gx0 = blockIdx.x * TW;
    const int gy0 = blockIdx.y * TH;

    const uint32_t sW_off   = (uint32_t)(uintptr_t)(&sW[0]);
    const uint32_t sFeat_off = (uint32_t)(uintptr_t)(&sFeat[0]);

    // Stage packed weights (NRECS KB) into LDS via async DMA.
    for (int i = tid; i < NRECS * 64; i += 256)
        async_copy16(sW_off + (uint32_t)i * 16u, (const uint4*)wpk + i);

    // Stage input tile + halo.
    const bool interior = (gx0 >= 1) && (gx0 + ROWPIX - 1 <= W_IMG) &&
                          (gy0 >= 1) && (gy0 + 5 <= H_IMG);
    if (interior) {
        const uint4* gbase =
            (const uint4*)(fin + (size_t)((gy0 - 1) * W_IMG + (gx0 - 1)) * CPACK);
        for (int i = tid; i < NCHUNK; i += 256) {
            int pi = i / CPP, part = i - pi * CPP;
            int s = pi / ROWPIX, t = pi - s * ROWPIX;
            async_copy16(sFeat_off + (uint32_t)i * 16u,
                         gbase + (size_t)(s * W_IMG + t) * CPP + part);
        }
    } else {
        uint4* dst = (uint4*)sFeat;
        for (int i = tid; i < NCHUNK; i += 256) {
            int pi = i / CPP, part = i - pi * CPP;
            int s = pi / ROWPIX, t = pi - s * ROWPIX;
            int gy = gy0 - 1 + s, gx = gx0 - 1 + t;
            if (gx >= 0 && gx < W_IMG && gy >= 0 && gy < H_IMG)
                async_copy16(sFeat_off + (uint32_t)i * 16u,
                             (const uint4*)fin + (size_t)(gy * W_IMG + gx) * CPP + part);
            else
                dst[i] = make_uint4(0u, 0u, 0u, 0u);
        }
    }
    async_wait0();            // this wave's async DMAs landed in LDS
    __syncthreads();          // all waves' staging visible

    const int wave = tid >> 5;
    const int lane = tid & 31;
    const int ry = wave >> 1;                  // output row within tile: 0..3
    const int chunk = (wave & 1) * 16;         // x offset of this wave's M-tile
    const int hi = lane >> 4;                  // lane half
    const int nl = lane & 15;

    v8f acc[NT];
#pragma unroll
    for (int nt = 0; nt < NT; ++nt) {
        float bv = bias[nt * 16 + nl];
#pragma unroll
        for (int i = 0; i < 8; ++i) acc[nt][i] = bv;
    }

#pragma unroll
    for (int dy = 0; dy < 3; ++dy) {
        const int s = ry + dy;                 // staged row (gy0-1+s == gy+dy-1)
#pragma unroll
        for (int kk = 0; kk < KK; ++kk) {
            // A fragment: pixel M = nl, K contiguous over (x-1..x+1, channels).
            const _Float16* arow =
                sFeat + s * (ROWPIX * CPACK) + (chunk + nl) * CPACK + kk * 32 + hi * 16;
            union { v16h v; v8h h[2]; } A;
            A.h[0] = *(const v8h*)(arow);
            A.h[1] = *(const v8h*)(arow + 8);
#pragma unroll
            for (int nt = 0; nt < NT; ++nt) {
                const _Float16* brec = sW + (((nt * 3 + dy) * KK + kk) << 9) + lane * 16;
                union { v16h v; v8h h[2]; } B;
                B.h[0] = *(const v8h*)(brec);
                B.h[1] = *(const v8h*)(brec + 8);
                acc[nt] = __builtin_amdgcn_wmma_f32_16x16x32_f16(
                    false, A.v, false, B.v, (short)0, acc[nt], false, false);
            }
        }
    }

    // Store: lane nl = output channel, acc VGPR r -> pixel r + 8*hi in M-tile.
    const int gy = gy0 + ry;
#pragma unroll
    for (int r = 0; r < 8; ++r) {
        int p = r + 8 * hi;
        int gx = gx0 + chunk + p;
        bool ok = (gx < W_IMG) && (gy < H_IMG);
#pragma unroll
        for (int nt = 0; nt < NT; ++nt) {
            float v = acc[nt][r];
            if (RELU) v = fmaxf(v, 0.f);
            int ch = nt * 16 + nl;
            if (ok && ch < COUTS)
                fout[(size_t)(gy * W_IMG + gx) * COUTS + ch] = (_Float16)v;
        }
    }
}

// ---------------------------------------------------------------------------
// Composite: out = clip(xb + strength * tanh(res) * sigmoid(attn), 0, 1)
// ---------------------------------------------------------------------------
__global__ void k_composite(const float4* __restrict__ xb,
                            const _Float16* __restrict__ fin4,
                            const float* __restrict__ strength_,
                            float* __restrict__ out) {
    int p = blockIdx.x * blockDim.x + threadIdx.x;
    if (p >= NPIX) return;
    float s = clampf(strength_[0], 0.02f, 0.2f);
    float4 b = xb[p];
    float rr = (float)fin4[p * 4 + 0];
    float rg = (float)fin4[p * 4 + 1];
    float rb = (float)fin4[p * 4 + 2];
    float at = (float)fin4[p * 4 + 3];
    float attn = 1.f / (1.f + expf(-at));
    out[p]            = clampf(b.x + s * tanhf(rr) * attn, 0.f, 1.f);
    out[NPIX + p]     = clampf(b.y + s * tanhf(rg) * attn, 0.f, 1.f);
    out[2 * NPIX + p] = clampf(b.z + s * tanhf(rb) * attn, 0.f, 1.f);
}

// ---------------------------------------------------------------------------
// Host launcher
// ---------------------------------------------------------------------------
extern "C" void kernel_launch(void* const* d_in, const int* in_sizes, int n_in,
                              void* d_out, int out_size, void* d_ws, size_t ws_size,
                              hipStream_t stream) {
    const float* x    = (const float*)d_in[0];
    const float* lut  = (const float*)d_in[1];
    const float* cm   = (const float*)d_in[2];
    const float* cb   = (const float*)d_in[3];
    const float* sh   = (const float*)d_in[4];
    const float* mi   = (const float*)d_in[5];
    const float* hl   = (const float*)d_in[6];
    const float* co   = (const float*)d_in[7];
    const float* sat  = (const float*)d_in[8];
    const float* wa   = (const float*)d_in[9];
    const float* lb   = (const float*)d_in[10];
    const float* rs   = (const float*)d_in[11];
    const float* w1   = (const float*)d_in[12];
    const float* bn1g = (const float*)d_in[13];
    const float* bn1b = (const float*)d_in[14];
    const float* bn1m = (const float*)d_in[15];
    const float* bn1v = (const float*)d_in[16];
    const float* w2   = (const float*)d_in[17];
    const float* bn2g = (const float*)d_in[18];
    const float* bn2b = (const float*)d_in[19];
    const float* bn2m = (const float*)d_in[20];
    const float* bn2v = (const float*)d_in[21];
    const float* w3   = (const float*)d_in[22];
    const float* bn3g = (const float*)d_in[23];
    const float* bn3b = (const float*)d_in[24];
    const float* bn3m = (const float*)d_in[25];
    const float* bn3v = (const float*)d_in[26];
    const float* wres = (const float*)d_in[27];
    const float* resb = (const float*)d_in[28];
    const float* watn = (const float*)d_in[29];
    const float* atnb = (const float*)d_in[30];
    float* out = (float*)d_out;

    // ---- workspace layout (all 256B aligned) ----
    uint8_t* ws = (uint8_t*)d_ws;
    size_t off = 0;
    auto alloc = [&](size_t bytes) -> uint8_t* {
        uint8_t* p = ws + off;
        off = (off + bytes + 255) & ~(size_t)255;
        return p;
    };
    float* sc1   = (float*)alloc(32 * 4);
    float* bias1 = (float*)alloc(32 * 4);
    float* sc2   = (float*)alloc(32 * 4);
    float* bias2 = (float*)alloc(32 * 4);
    float* sc3   = (float*)alloc(16 * 4);
    float* bias3 = (float*)alloc(16 * 4);
    float* biasF = (float*)alloc(16 * 4);
    _Float16* wpk1 = (_Float16*)alloc(6 * 1024);   // NT=2 KK=1
    _Float16* wpk2 = (_Float16*)alloc(18 * 1024);  // NT=2 KK=3
    _Float16* wpk3 = (_Float16*)alloc(9 * 1024);   // NT=1 KK=3
    _Float16* wpkF = (_Float16*)alloc(6 * 1024);   // NT=1 KK=2
    float4*   xb   = (float4*)alloc((size_t)NPIX * 16);
    v8h*      xin  = (v8h*)alloc((size_t)NPIX * 16);
    _Float16* fA   = (_Float16*)alloc((size_t)NPIX * 64);  // 32ch f16
    _Float16* fB   = (_Float16*)alloc((size_t)NPIX * 64);  // 32ch f16
    _Float16* f3   = fA;                                    // 16ch, reuse fA
    _Float16* fin4 = fB;                                    // 4ch, reuse fB

    // ---- prep: BN fold + weight packing ----
    k_fold_bn<<<1, 64, 0, stream>>>(bn1g, bn1b, bn1m, bn1v, sc1, bias1, 32, 32);
    k_fold_bn<<<1, 64, 0, stream>>>(bn2g, bn2b, bn2m, bn2v, sc2, bias2, 32, 32);
    k_fold_bn<<<1, 64, 0, stream>>>(bn3g, bn3b, bn3m, bn3v, sc3, bias3, 16, 16);
    k_final_bias<<<1, 32, 0, stream>>>(resb, atnb, biasF);

    k_pack_w<<<(2 * 3 * 1 * 512 + 255) / 256, 256, 0, stream>>>(w1, sc1, wpk1, 6, 32, 8, 2, 1);
    k_pack_w<<<(2 * 3 * 3 * 512 + 255) / 256, 256, 0, stream>>>(w2, sc2, wpk2, 32, 32, 32, 2, 3);
    k_pack_w<<<(1 * 3 * 3 * 512 + 255) / 256, 256, 0, stream>>>(w3, sc3, wpk3, 32, 16, 32, 1, 3);
    k_pack_w_final<<<(1 * 3 * 2 * 512 + 255) / 256, 256, 0, stream>>>(wres, watn, wpkF);

    // ---- pointwise + LUT ----
    k_pointwise_lut<<<(NPIX + 255) / 256, 256, 0, stream>>>(
        x, lut, cm, cb, sh, mi, hl, co, sat, wa, lb, xb, xin);

    // ---- conv stack (WMMA) ----
    dim3 grid((W_IMG + 31) / 32, (H_IMG + 3) / 4);
    k_conv3x3_wmma<8, 2, 1, 32, true><<<grid, 256, 0, stream>>>(
        (const _Float16*)xin, wpk1, bias1, fA);
    k_conv3x3_wmma<32, 2, 3, 32, true><<<grid, 256, 0, stream>>>(fA, wpk2, bias2, fB);
    k_conv3x3_wmma<32, 1, 3, 16, true><<<grid, 256, 0, stream>>>(fB, wpk3, bias3, f3);
    k_conv3x3_wmma<16, 1, 2, 4, false><<<grid, 256, 0, stream>>>(f3, wpkF, biasF, fin4);

    // ---- composite ----
    k_composite<<<(NPIX + 255) / 256, 256, 0, stream>>>(xb, fin4, rs, out);
}
